// ScaledDotProductAttention_81243601371700
// MI455X (gfx1250) — matrix operand order repaired and checked by reading, not compiled, not run on hardware
//
#include <hip/hip_runtime.h>
#include <hip/hip_bf16.h>

typedef __attribute__((ext_vector_type(16))) _Float16 v16h;
typedef __attribute__((ext_vector_type(8)))  float    v8f;

#define B_   4
#define H_   16
#define S_   2048
#define DK_  128
#define KSTEP 32
#define QB    128   // q rows per block (8 waves * 16)
#define SCALE 0.08838834764831845f  // 1/sqrt(128)

// A-fragment (16x32 f16) per-lane K offset: vgpr i holds pair (k, k+1)
__device__ __forceinline__ int koffA(int i, int half) {
    return (i < 4) ? (2 * i + 8 * half) : (16 + 2 * (i - 4) + 8 * half);
}
// B-fragment (32x16 f16) per-lane K offset: vgpr i holds pair (k, k+1)
__device__ __forceinline__ int koffB(int i, int half) {
    return (i < 4) ? (2 * i + 8 * half) : (16 + 2 * (i - 4) + 8 * half);
}

__global__ __launch_bounds__(256)
void attn_fwd_kernel(const float* __restrict__ q, const float* __restrict__ k,
                     const float* __restrict__ v, const int* __restrict__ mask,
                     float* __restrict__ out, float* __restrict__ weights) {
    const int bh   = blockIdx.y;
    const int qblk = blockIdx.x;

    const float* qp = q + (size_t)bh * S_ * DK_;
    const float* kp = k + (size_t)bh * S_ * DK_;
    const float* vp = v + (size_t)bh * S_ * DK_;
    const int*   mp = mask + (size_t)bh * S_ * S_;
    float*       wp = weights + (size_t)bh * S_ * S_;
    float*       op = out + (size_t)bh * S_ * DK_;

    const int tid  = threadIdx.x;
    const int wave = tid >> 5;
    const int lane = tid & 31;
    const int m16  = lane & 15;
    const int half = lane >> 4;
    const int qrow0 = qblk * QB + wave * 16;

    __shared__ _Float16 Klds[KSTEP][DK_];        // 8 KB: K tile, row-major [krow][feat]
    __shared__ _Float16 VldsT[DK_][KSTEP + 2];   // ~8.5 KB: V tile transposed [feat][krow]
    __shared__ _Float16 Plds[8][16][KSTEP];      // 8 KB: per-wave P staging [qrow][kcol]

    // ---- Q A-fragments (16x32 f16), 4 chunks covering DK=128 ----
    v16h qa[4];
#pragma unroll
    for (int c = 0; c < 4; ++c) {
#pragma unroll
        for (int i = 0; i < 8; ++i) {
            int kf = 32 * c + koffA(i, half);
            const float* src = qp + (size_t)(qrow0 + m16) * DK_ + kf;
            qa[c][2 * i]     = (_Float16)src[0];
            qa[c][2 * i + 1] = (_Float16)src[1];
        }
    }

    v8f o[8];
#pragma unroll
    for (int dt = 0; dt < 8; ++dt) o[dt] = (v8f){0,0,0,0,0,0,0,0};
    float mrow[8], lrow[8];
#pragma unroll
    for (int j = 0; j < 8; ++j) { mrow[j] = -1e30f; lrow[j] = 0.0f; }

    for (int kb = 0; kb < S_; kb += KSTEP) {
        __syncthreads();
        // ---- cooperative stage K (row-major) and V (transposed) to LDS as f16 ----
#pragma unroll
        for (int r = 0; r < 4; ++r) {
            int idx = r * 256 + tid;      // 1024 float4 per tile
            int kk  = idx >> 5;           // 32 float4 per row
            int f   = (idx & 31) * 4;
            const float4 kv4 = *(const float4*)(kp + (size_t)(kb + kk) * DK_ + f);
            Klds[kk][f + 0] = (_Float16)kv4.x;
            Klds[kk][f + 1] = (_Float16)kv4.y;
            Klds[kk][f + 2] = (_Float16)kv4.z;
            Klds[kk][f + 3] = (_Float16)kv4.w;
            const float4 vv4 = *(const float4*)(vp + (size_t)(kb + kk) * DK_ + f);
            VldsT[f + 0][kk] = (_Float16)vv4.x;
            VldsT[f + 1][kk] = (_Float16)vv4.y;
            VldsT[f + 2][kk] = (_Float16)vv4.z;
            VldsT[f + 3][kk] = (_Float16)vv4.w;
        }
        __syncthreads();

        // ---- scores: two 16x16 tiles (t=0,1) over 32 k columns ----
        v8f c0 = (v8f){0,0,0,0,0,0,0,0};
        v8f c1 = (v8f){0,0,0,0,0,0,0,0};
#pragma unroll
        for (int cc = 0; cc < 4; ++cc) {
            v16h b0, b1;
#pragma unroll
            for (int i = 0; i < 8; ++i) {
                int kf = 32 * cc + koffB(i, half);
                b0[2 * i]     = Klds[m16][kf];
                b0[2 * i + 1] = Klds[m16][kf + 1];
                b1[2 * i]     = Klds[16 + m16][kf];
                b1[2 * i + 1] = Klds[16 + m16][kf + 1];
            }
            c0 = __builtin_amdgcn_wmma_f32_16x16x32_f16(false, qa[cc], false, b0,
                                                        (short)0, c0, false, false);
            c1 = __builtin_amdgcn_wmma_f32_16x16x32_f16(false, qa[cc], false, b1,
                                                        (short)0, c1, false, false);
        }

        // ---- mask + scale + online softmax (rows live in 16-lane groups) ----
#pragma unroll
        for (int j = 0; j < 8; ++j) {
            int row = j + 8 * half;
            int qr  = qrow0 + row;
            float s0 = c0[j] * SCALE;
            float s1 = c1[j] * SCALE;
            int mk0 = mp[(size_t)qr * S_ + kb + m16];
            int mk1 = mp[(size_t)qr * S_ + kb + 16 + m16];
            wp[(size_t)qr * S_ + kb + m16]      = mk0 ? s0 : -__builtin_inff();
            wp[(size_t)qr * S_ + kb + 16 + m16] = mk1 ? s1 : -__builtin_inff();
            float a0 = mk0 ? s0 : -1e30f;
            float a1 = mk1 ? s1 : -1e30f;
            float tm = fmaxf(a0, a1);
            tm = fmaxf(tm, __shfl_xor(tm, 1, 32));
            tm = fmaxf(tm, __shfl_xor(tm, 2, 32));
            tm = fmaxf(tm, __shfl_xor(tm, 4, 32));
            tm = fmaxf(tm, __shfl_xor(tm, 8, 32));
            float mnew  = fmaxf(mrow[j], tm);
            float alpha = __expf(mrow[j] - mnew);
            float p0 = __expf(a0 - mnew);
            float p1 = __expf(a1 - mnew);
            float ps = p0 + p1;
            ps += __shfl_xor(ps, 1, 32);
            ps += __shfl_xor(ps, 2, 32);
            ps += __shfl_xor(ps, 4, 32);
            ps += __shfl_xor(ps, 8, 32);
            lrow[j] = lrow[j] * alpha + ps;
            mrow[j] = mnew;
#pragma unroll
            for (int dt = 0; dt < 8; ++dt) o[dt][j] *= alpha;
            Plds[wave][row][m16]      = (_Float16)p0;
            Plds[wave][row][16 + m16] = (_Float16)p1;
        }
        asm volatile("s_wait_dscnt 0" ::: "memory");

        // ---- P A-fragment (16x32) from per-wave staging ----
        v16h pa;
#pragma unroll
        for (int i = 0; i < 8; ++i) {
            int kf = koffA(i, half);
            pa[2 * i]     = Plds[wave][m16][kf];
            pa[2 * i + 1] = Plds[wave][m16][kf + 1];
        }
        // ---- O += P * V for 8 DK tiles ----
#pragma unroll
        for (int dt = 0; dt < 8; ++dt) {
            v16h bv;
#pragma unroll
            for (int i = 0; i < 8; ++i) {
                int kf = koffB(i, half);
                bv[2 * i]     = VldsT[dt * 16 + m16][kf];
                bv[2 * i + 1] = VldsT[dt * 16 + m16][kf + 1];
            }
            o[dt] = __builtin_amdgcn_wmma_f32_16x16x32_f16(false, pa, false, bv,
                                                           (short)0, o[dt], false, false);
        }
    }

    // ---- finalize: O /= l, write out ----
#pragma unroll
    for (int j = 0; j < 8; ++j) {
        float inv = 1.0f / lrow[j];
#pragma unroll
        for (int dt = 0; dt < 8; ++dt) o[dt][j] *= inv;
    }
#pragma unroll
    for (int dt = 0; dt < 8; ++dt) {
#pragma unroll
        for (int j = 0; j < 8; ++j) {
            int row = j + 8 * half;
            op[(size_t)(qrow0 + row) * DK_ + dt * 16 + m16] = o[dt][j];
        }
    }
}

// Pass B: per-row streaming softmax of the raw masked scores, in place.
__global__ __launch_bounds__(256)
void softmax_rows_kernel(float* __restrict__ weights) {
    float* row = weights + (size_t)blockIdx.x * S_;
    const int tid = threadIdx.x;
    __shared__ float red[256];

    float vbuf[8];
#pragma unroll
    for (int i = 0; i < 8; ++i) vbuf[i] = row[tid + i * 256];

    float mx = vbuf[0];
#pragma unroll
    for (int i = 1; i < 8; ++i) mx = fmaxf(mx, vbuf[i]);
    red[tid] = mx;
    __syncthreads();
    for (int s = 128; s > 0; s >>= 1) {
        if (tid < s) red[tid] = fmaxf(red[tid], red[tid + s]);
        __syncthreads();
    }
    mx = red[0];
    __syncthreads();

    float sum = 0.0f;
#pragma unroll
    for (int i = 0; i < 8; ++i) {
        vbuf[i] = __expf(vbuf[i] - mx);
        sum += vbuf[i];
    }
    red[tid] = sum;
    __syncthreads();
    for (int s = 128; s > 0; s >>= 1) {
        if (tid < s) red[tid] += red[tid + s];
        __syncthreads();
    }
    sum = red[0];

    float inv = 1.0f / sum;
#pragma unroll
    for (int i = 0; i < 8; ++i) row[tid + i * 256] = vbuf[i] * inv;
}

extern "C" void kernel_launch(void* const* d_in, const int* in_sizes, int n_in,
                              void* d_out, int out_size, void* d_ws, size_t ws_size,
                              hipStream_t stream) {
    (void)in_sizes; (void)n_in; (void)out_size; (void)d_ws; (void)ws_size;
    const float* q    = (const float*)d_in[0];
    const float* k    = (const float*)d_in[1];
    const float* v    = (const float*)d_in[2];
    const int*   mask = (const int*)d_in[3];
    float* out     = (float*)d_out;
    float* weights = out + (size_t)B_ * H_ * S_ * DK_;

    dim3 gridA(S_ / QB, B_ * H_);
    attn_fwd_kernel<<<gridA, 256, 0, stream>>>(q, k, v, mask, out, weights);

    dim3 gridB(B_ * H_ * S_);
    softmax_rows_kernel<<<gridB, 256, 0, stream>>>(weights);
}
